// model_encdec_GCN_38311108280383
// MI455X (gfx1250) — matrix operand order repaired and checked by reading, compiled, not run on hardware
//
#include <hip/hip_runtime.h>
#include <hip/hip_bf16.h>
#include <math.h>

// ---------------------------------------------------------------- constants
#define KCH    48
#define PASTL  20
#define FUTL   40
#define NNODES 65536
#define NEDGES 1048576

typedef __bf16 v16bf __attribute__((ext_vector_type(16)));
typedef float  v8f   __attribute__((ext_vector_type(8)));
typedef int    v4i_  __attribute__((ext_vector_type(4)));

union BfFrag {
  v16bf bf;
  uint4 q[2];
  unsigned short u16[16];
};

__device__ __forceinline__ unsigned short f2bf(float f) {
  unsigned int u = __float_as_uint(f);
  u += 0x7FFFu + ((u >> 16) & 1u);           // round-to-nearest-even
  return (unsigned short)(u >> 16);
}
__device__ __forceinline__ float sigmoidf_(float x) {
  return 1.0f / (1.0f + __expf(-x));
}

// A-fragment (16x32 bf16, wave32): lane holds row m=lane&15.
// low lanes: K = c*32 + {0..7, 16..23};  high lanes: K = c*32 + {8..15, 24..31}
__device__ __forceinline__ v16bf load_a_frag(const unsigned short* base /*[16][64]*/,
                                             int lane, int chunk) {
  int m  = lane & 15;
  int hb = (lane >> 4) & 1;
  const unsigned short* p = base + m * 64 + chunk * 32 + hb * 8;
  BfFrag f;
  f.q[0] = *(const uint4*)(p);
  f.q[1] = *(const uint4*)(p + 16);
  return f.bf;
}
// B-fragment: prearranged contiguous 16 bf16 per lane
__device__ __forceinline__ v16bf load_b_frag(const unsigned short* p) {
  BfFrag f;
  f.q[0] = *(const uint4*)(p);
  f.q[1] = *(const uint4*)(p + 8);
  return f.bf;
}

// Async global->LDS DMA if this toolchain exposes it (gfx1250 path).
#if defined(__HIP_DEVICE_COMPILE__) && \
    __has_builtin(__builtin_amdgcn_global_load_async_to_lds_b128)
#define HAVE_ASYNC_LDS 1
typedef __attribute__((address_space(1))) v4i_* async_gp_t;
typedef __attribute__((address_space(3))) v4i_* async_lp_t;
#endif

// ============================================================== encoder
// Fused conv1d(K=3,pad=1)+ReLU + 20-step GRU(hidden 48) over 65536 nodes.
// 2 waves/block, 16 nodes/wave, gates via v_wmma_f32_16x16x32_bf16.
#define ENC_WAVES 2
__global__ __launch_bounds__(64)
void encode_nodes_kernel(const float* __restrict__ x,      // (N,20,2)
                         const float* __restrict__ convw,  // (48,2,3)
                         const float* __restrict__ convb,  // (48)
                         const float* __restrict__ Wih,    // (144,48)
                         const float* __restrict__ Whh,    // (144,48)
                         const float* __restrict__ bih,    // (144)
                         const float* __restrict__ bhh,    // (144)
                         float* __restrict__ state_out)    // (N,48)
{
  __shared__ __align__(16) unsigned short sBih[9][2][32][16];   // 18 KB
  __shared__ __align__(16) unsigned short sBhh[9][2][32][16];   // 18 KB
  __shared__ float sConvW[288];
  __shared__ float sConvB[48];
  __shared__ float sBi[144], sBh[144];
  __shared__ __align__(16) unsigned short sEmb[ENC_WAVES][16][64]; // bf16, K padded
  __shared__ __align__(16) unsigned short sH[ENC_WAVES][16][64];
  __shared__ __align__(16) float sX[ENC_WAVES][16][PASTL * 2];     // staged inputs

  const int tid  = threadIdx.x;
  const int wv   = tid >> 5, lane = tid & 31;
  const int nodeBase = blockIdx.x * (ENC_WAVES * 16) + wv * 16;

  // ---- stage this wave's 16 node trajectories into LDS (async DMA if present)
#ifdef HAVE_ASYNC_LDS
  for (int idx = lane; idx < 16 * 10; idx += 32) {        // 10 x b128 per node
    int m = idx / 10, c4 = idx % 10;
    const float* g = x + (size_t)(nodeBase + m) * (PASTL * 2) + c4 * 4;
    __builtin_amdgcn_global_load_async_to_lds_b128(
        (async_gp_t)(g), (async_lp_t)(&sX[wv][m][c4 * 4]), 0, 0);
  }
  asm volatile("s_wait_asynccnt 0" ::: "memory");
#else
  for (int idx = lane; idx < 16 * PASTL * 2; idx += 32) {
    int m = idx / (PASTL * 2), r = idx % (PASTL * 2);
    sX[wv][m][r] = x[(size_t)(nodeBase + m) * (PASTL * 2) + r];
  }
#endif

  // ---- pre-swizzle B = W^T (48K x 144N) into WMMA B-fragments (bf16)
  for (int idx = tid; idx < 9 * 2 * 32 * 16; idx += blockDim.x) {
    int i = idx & 15, l = (idx >> 4) & 31, c = (idx >> 9) & 1, n = idx >> 10;
    int col = n * 16 + (l & 15);
    int k   = c * 32 + ((l & 16) ? 8 : 0) + (i < 8 ? i : i + 8);
    unsigned short vi = 0, vh = 0;
    if (k < KCH) { vi = f2bf(Wih[col * KCH + k]); vh = f2bf(Whh[col * KCH + k]); }
    sBih[n][c][l][i] = vi;
    sBhh[n][c][l][i] = vh;
  }
  for (int i = tid; i < 288; i += blockDim.x) sConvW[i] = convw[i];
  for (int i = tid; i < 48;  i += blockDim.x) sConvB[i] = convb[i];
  for (int i = tid; i < 144; i += blockDim.x) { sBi[i] = bih[i]; sBh[i] = bhh[i]; }

  // zero h (incl. K padding) and embed padding
  for (int idx = lane; idx < 16 * 64; idx += 32) sH[wv][idx >> 6][idx & 63] = 0;
  for (int idx = lane; idx < 16 * 16; idx += 32) sEmb[wv][idx >> 4][48 + (idx & 15)] = 0;
  __syncthreads();

  float hc[3][8];
  #pragma unroll
  for (int j = 0; j < 3; ++j)
    #pragma unroll
    for (int r = 0; r < 8; ++r) hc[j][r] = 0.f;

  const int cl = lane & 15, hi = (lane >> 4) & 1;

  for (int t = 0; t < PASTL; ++t) {
    // conv1d + relu -> bf16 embeds in A-fragment-friendly layout
    for (int idx = lane; idx < 16 * KCH; idx += 32) {
      int m = idx / KCH, f = idx % KCH;
      float a = sConvB[f];
      #pragma unroll
      for (int tap = 0; tap < 3; ++tap) {
        int tt = t - 1 + tap;
        if (tt >= 0 && tt < PASTL) {
          a += sConvW[f * 6 + tap]     * sX[wv][m][tt * 2 + 0];
          a += sConvW[f * 6 + 3 + tap] * sX[wv][m][tt * 2 + 1];
        }
      }
      sEmb[wv][m][f] = f2bf(fmaxf(a, 0.f));
    }
    asm volatile("s_wait_dscnt 0" ::: "memory");

    v16bf aE0 = load_a_frag(&sEmb[wv][0][0], lane, 0);
    v16bf aE1 = load_a_frag(&sEmb[wv][0][0], lane, 1);
    v16bf aH0 = load_a_frag(&sH[wv][0][0],   lane, 0);
    v16bf aH1 = load_a_frag(&sH[wv][0][0],   lane, 1);

    v8f gi[9], gh[9];
    #pragma unroll
    for (int n = 0; n < 9; ++n) {
      v8f a = {};
      a = __builtin_amdgcn_wmma_f32_16x16x32_bf16(false, aE0, false,
            load_b_frag(&sBih[n][0][lane][0]), (short)0, a, false, false);
      a = __builtin_amdgcn_wmma_f32_16x16x32_bf16(false, aE1, false,
            load_b_frag(&sBih[n][1][lane][0]), (short)0, a, false, false);
      gi[n] = a;
      v8f b = {};
      b = __builtin_amdgcn_wmma_f32_16x16x32_bf16(false, aH0, false,
            load_b_frag(&sBhh[n][0][lane][0]), (short)0, b, false, false);
      b = __builtin_amdgcn_wmma_f32_16x16x32_bf16(false, aH1, false,
            load_b_frag(&sBhh[n][1][lane][0]), (short)0, b, false, false);
      gh[n] = b;
    }

    // GRU gates (accumulators are in C-layout: lane/VGPR -> (row, col))
    #pragma unroll
    for (int j = 0; j < 3; ++j) {
      int c = 16 * j + cl;
      float bir = sBi[c],       bhr = sBh[c];
      float biz = sBi[48 + c],  bhz = sBh[48 + c];
      float bin = sBi[96 + c],  bhn = sBh[96 + c];
      #pragma unroll
      for (int r = 0; r < 8; ++r) {
        float rg = sigmoidf_(gi[j][r]     + bir + gh[j][r]     + bhr);
        float zg = sigmoidf_(gi[j + 3][r] + biz + gh[j + 3][r] + bhz);
        float hn = gh[j + 6][r] + bhn;
        float nn = tanhf(gi[j + 6][r] + bin + rg * hn);
        float hv = (1.f - zg) * nn + zg * hc[j][r];
        hc[j][r] = hv;
        int m = r + 8 * hi;
        sH[wv][m][c] = f2bf(hv);
      }
    }
    asm volatile("s_wait_dscnt 0" ::: "memory");
  }

  // write final hidden state (f32)
  #pragma unroll
  for (int j = 0; j < 3; ++j) {
    int c = 16 * j + cl;
    #pragma unroll
    for (int r = 0; r < 8; ++r) {
      int m = r + 8 * hi;
      state_out[(size_t)(nodeBase + m) * KCH + c] = hc[j][r];
    }
  }
}

// ============================================================== Y = X @ W^T (48x48) via WMMA
__global__ __launch_bounds__(128)
void matmul48_kernel(const float* __restrict__ X, const float* __restrict__ W,
                     float* __restrict__ Y)
{
  __shared__ __align__(16) unsigned short sB[3][2][32][16];
  __shared__ __align__(16) unsigned short sA[4][16][64];
  const int tid = threadIdx.x;
  for (int idx = tid; idx < 3 * 2 * 32 * 16; idx += blockDim.x) {
    int i = idx & 15, l = (idx >> 4) & 31, c = (idx >> 9) & 1, n = idx >> 10;
    int col = n * 16 + (l & 15);
    int k   = c * 32 + ((l & 16) ? 8 : 0) + (i < 8 ? i : i + 8);
    sB[n][c][l][i] = (k < KCH) ? f2bf(W[col * KCH + k]) : (unsigned short)0;
  }
  const int wv = tid >> 5, lane = tid & 31;
  const int rowBase = blockIdx.x * 64 + wv * 16;
  for (int idx = lane; idx < 16 * 64; idx += 32) {
    int m = idx >> 6, k = idx & 63;
    sA[wv][m][k] = (k < KCH) ? f2bf(X[(size_t)(rowBase + m) * KCH + k]) : (unsigned short)0;
  }
  __syncthreads();

  v16bf a0 = load_a_frag(&sA[wv][0][0], lane, 0);
  v16bf a1 = load_a_frag(&sA[wv][0][0], lane, 1);
  const int cl = lane & 15, hi = (lane >> 4) & 1;
  #pragma unroll
  for (int n = 0; n < 3; ++n) {
    v8f acc = {};
    acc = __builtin_amdgcn_wmma_f32_16x16x32_bf16(false, a0, false,
            load_b_frag(&sB[n][0][lane][0]), (short)0, acc, false, false);
    acc = __builtin_amdgcn_wmma_f32_16x16x32_bf16(false, a1, false,
            load_b_frag(&sB[n][1][lane][0]), (short)0, acc, false, false);
    #pragma unroll
    for (int r = 0; r < 8; ++r) {
      int m = r + 8 * hi;
      Y[(size_t)(rowBase + m) * KCH + 16 * n + cl] = acc[r];
    }
  }
}

// ============================================================== graph helpers
__global__ void zero_kernel(float* __restrict__ p, int n) {
  int i = blockIdx.x * blockDim.x + threadIdx.x;
  if (i < n) p[i] = 0.f;
}

__global__ void deg_kernel(const int* __restrict__ ei, const float* __restrict__ ew,
                           float* __restrict__ deg) {
  int e = blockIdx.x * blockDim.x + threadIdx.x;
  if (e < NEDGES) atomicAdd(&deg[ei[NEDGES + e]], ew[e]);
}

__global__ void dinv_kernel(const float* __restrict__ deg, float* __restrict__ dinv) {
  int i = blockIdx.x * blockDim.x + threadIdx.x;
  if (i < NNODES) dinv[i] = rsqrtf(deg[i] + 1.0f);   // +1 = self loop weight
}

__global__ void scatter_kernel(const int* __restrict__ ei, const float* __restrict__ ew,
                               const float* __restrict__ dinv, const float* __restrict__ xw,
                               float* __restrict__ agg) {
  int e = blockIdx.x * blockDim.x + threadIdx.x;
  if (e >= NEDGES) return;
  __builtin_prefetch(&ei[e + 2048], 0, 0);
  int row = ei[e], col = ei[NEDGES + e];
  float norm = dinv[row] * ew[e] * dinv[col];
  const float4* src = (const float4*)(xw + (size_t)row * KCH);
  float* dst = agg + (size_t)col * KCH;
  #pragma unroll
  for (int q = 0; q < 12; ++q) {
    float4 v = src[q];
    atomicAdd(dst + q * 4 + 0, v.x * norm);
    atomicAdd(dst + q * 4 + 1, v.y * norm);
    atomicAdd(dst + q * 4 + 2, v.z * norm);
    atomicAdd(dst + q * 4 + 3, v.w * norm);
  }
}

// x1 = relu(agg + xw*dinv^2 (self loop) + b)   (in place on agg)
__global__ void finish1_kernel(float* __restrict__ agg, const float* __restrict__ xw,
                               const float* __restrict__ dinv, const float* __restrict__ b1) {
  int idx = blockIdx.x * blockDim.x + threadIdx.x;
  if (idx >= NNODES * KCH) return;
  int node = idx / KCH, c = idx % KCH;
  float di = dinv[node];
  agg[idx] = fmaxf(agg[idx] + xw[idx] * di * di + b1[c], 0.f);
}

// layer-2: only node 0's aggregation is needed downstream
__global__ void gather0_kernel(const int* __restrict__ ei, const float* __restrict__ ew,
                               const float* __restrict__ dinv, const float* __restrict__ xw2,
                               float* __restrict__ interact /*48, pre-zeroed*/) {
  int e = blockIdx.x * blockDim.x + threadIdx.x;
  if (e >= NEDGES) return;
  if (ei[NEDGES + e] != 0) return;
  int row = ei[e];
  float norm = dinv[row] * ew[e] * dinv[0];
  const float* src = xw2 + (size_t)row * KCH;
  for (int c = 0; c < KCH; ++c) atomicAdd(&interact[c], src[c] * norm);
}

// ============================================================== tiny B=1 sequence kernels
__device__ void gru_seq_small(const float* seq, int len,
                              const float* cw, const float* cb,
                              const float* Wih, const float* Whh,
                              const float* bi, const float* bh,
                              float* h, float* emb, float* g) {
  int tid = threadIdx.x;
  if (tid < 48) h[tid] = 0.f;
  __syncthreads();
  for (int t = 0; t < len; ++t) {
    if (tid < 48) {
      float a = cb[tid];
      #pragma unroll
      for (int tap = 0; tap < 3; ++tap) {
        int tt = t - 1 + tap;
        if (tt >= 0 && tt < len) {
          a += cw[tid * 6 + tap]     * seq[tt * 2 + 0];
          a += cw[tid * 6 + 3 + tap] * seq[tt * 2 + 1];
        }
      }
      emb[tid] = fmaxf(a, 0.f);
    }
    __syncthreads();
    if (tid < 144) {
      float a = bi[tid], b = bh[tid];
      for (int k = 0; k < 48; ++k) {
        a += Wih[tid * 48 + k] * emb[k];
        b += Whh[tid * 48 + k] * h[k];
      }
      g[tid] = a; g[144 + tid] = b;
    }
    __syncthreads();
    if (tid < 48) {
      float r = sigmoidf_(g[tid]      + g[144 + tid]);
      float z = sigmoidf_(g[48 + tid] + g[192 + tid]);
      float n = tanhf(g[96 + tid] + r * g[240 + tid]);
      h[tid] = (1.f - z) * n + z * h[tid];
    }
    __syncthreads();
  }
}

__global__ __launch_bounds__(160)
void small_enc_kernel(const float* __restrict__ past, const float* __restrict__ future,
                      const float* cpw, const float* cpb,
                      const float* cfw, const float* cfb,
                      const float* pWih, const float* pWhh, const float* pbi, const float* pbh,
                      const float* fWih, const float* fWhh, const float* fbi, const float* fbh,
                      float* __restrict__ target_past, float* __restrict__ state_fut,
                      float* __restrict__ present0) {
  __shared__ float h[48], emb[48], g[288], seq[FUTL * 2];
  int tid = threadIdx.x;

  for (int i = tid; i < PASTL * 2; i += blockDim.x) seq[i] = past[i];
  __syncthreads();
  gru_seq_small(seq, PASTL, cpw, cpb, pWih, pWhh, pbi, pbh, h, emb, g);
  if (tid < 48) target_past[tid] = h[tid];
  if (tid < 2)  present0[tid] = past[(PASTL - 1) * 2 + tid];
  __syncthreads();

  for (int i = tid; i < FUTL * 2; i += blockDim.x) seq[i] = future[i];
  __syncthreads();
  gru_seq_small(seq, FUTL, cfw, cfb, fWih, fWhh, fbi, fbh, h, emb, g);
  if (tid < 48) state_fut[tid] = h[tid];
}

__global__ __launch_bounds__(448)
void decoder_kernel(const float* __restrict__ interact_raw, const float* __restrict__ xw2,
                    const float* __restrict__ dinv, const float* __restrict__ g2b,
                    const float* __restrict__ target, const float* __restrict__ fut,
                    const float* __restrict__ present0,
                    const float* __restrict__ Wih, const float* __restrict__ Whh,
                    const float* __restrict__ bi, const float* __restrict__ bh,
                    const float* __restrict__ fcw, const float* __restrict__ fcb,
                    float* __restrict__ out) {
  __shared__ float conc[144], h[144], SI[432], SG[432], pres[2];
  int tid = threadIdx.x;
  if (tid < 48) {
    float d0 = dinv[0];
    conc[tid]      = interact_raw[tid] + xw2[tid] * d0 * d0 + g2b[tid]; // gcn2 @ node0
    conc[48 + tid] = target[tid];
    conc[96 + tid] = fut[tid];
  }
  if (tid < 144) h[tid] = 0.f;
  if (tid < 2) pres[tid] = present0[tid];
  __syncthreads();
  if (tid < 432) {                       // gi at t=0 (inputs are zero for t>0)
    float a = bi[tid];
    for (int k = 0; k < 144; ++k) a += Wih[tid * 144 + k] * conc[k];
    SI[tid] = a;
  }
  __syncthreads();
  for (int t = 0; t < FUTL; ++t) {
    if (tid < 432) {
      float b = bh[tid];
      for (int k = 0; k < 144; ++k) b += Whh[tid * 144 + k] * h[k];
      SG[tid] = b;
    }
    __syncthreads();
    if (tid < 144) {
      float gi_r = (t == 0) ? SI[tid]       : bi[tid];
      float gi_z = (t == 0) ? SI[144 + tid] : bi[144 + tid];
      float gi_n = (t == 0) ? SI[288 + tid] : bi[288 + tid];
      float r = sigmoidf_(gi_r + SG[tid]);
      float z = sigmoidf_(gi_z + SG[144 + tid]);
      float n = tanhf(gi_n + r * SG[288 + tid]);
      h[tid] = (1.f - z) * n + z * h[tid];
    }
    __syncthreads();
    if (tid < 2) {
      float a = fcb[tid];
      for (int k = 0; k < 144; ++k) a += fcw[tid * 144 + k] * h[k];
      float c = pres[tid] + a;
      pres[tid] = c;
      out[t * 2 + tid] = c;
    }
    __syncthreads();
  }
}

// ============================================================== launch
extern "C" void kernel_launch(void* const* d_in, const int* in_sizes, int n_in,
                              void* d_out, int out_size, void* d_ws, size_t ws_size,
                              hipStream_t stream) {
  const float* past   = (const float*)d_in[0];
  const float* future = (const float*)d_in[1];
  const float* x      = (const float*)d_in[2];
  const int*   ei     = (const int*)  d_in[3];
  const float* ew     = (const float*)d_in[4];
  const float* cpw = (const float*)d_in[5],  *cpb = (const float*)d_in[6];
  const float* cfw = (const float*)d_in[7],  *cfb = (const float*)d_in[8];
  const float* pWih = (const float*)d_in[9],  *pWhh = (const float*)d_in[10];
  const float* pbi  = (const float*)d_in[11], *pbh  = (const float*)d_in[12];
  const float* fWih = (const float*)d_in[13], *fWhh = (const float*)d_in[14];
  const float* fbi  = (const float*)d_in[15], *fbh  = (const float*)d_in[16];
  const float* dWih = (const float*)d_in[17], *dWhh = (const float*)d_in[18];
  const float* dbi  = (const float*)d_in[19], *dbh  = (const float*)d_in[20];
  const float* fcw  = (const float*)d_in[21], *fcb  = (const float*)d_in[22];
  const float* g1w  = (const float*)d_in[23], *g1b  = (const float*)d_in[24];
  const float* g2w  = (const float*)d_in[25], *g2b  = (const float*)d_in[26];

  // workspace layout (~38.3 MB)
  float* state    = (float*)d_ws;                 // N*48
  float* xw       = state + (size_t)NNODES * KCH; // N*48 (xw, later xw2)
  float* agg      = xw    + (size_t)NNODES * KCH; // N*48 (agg -> x1 in place)
  float* deg      = agg   + (size_t)NNODES * KCH; // N
  float* dinv     = deg   + NNODES;               // N
  float* interact = dinv  + NNODES;               // 48
  float* target   = interact + 48;                // 48
  float* sfut     = target   + 48;                // 48
  float* pres0    = sfut     + 48;                // 2
  float* outf     = (float*)d_out;

  // 1) per-node conv+GRU encoder (WMMA, async-LDS staging)
  encode_nodes_kernel<<<NNODES / (ENC_WAVES * 16), 64, 0, stream>>>(
      x, cpw, cpb, pWih, pWhh, pbi, pbh, state);

  // 2) degrees / normalization
  zero_kernel<<<(NNODES + 255) / 256, 256, 0, stream>>>(deg, NNODES);
  deg_kernel<<<NEDGES / 256, 256, 0, stream>>>(ei, ew, deg);
  dinv_kernel<<<(NNODES + 255) / 256, 256, 0, stream>>>(deg, dinv);

  // 3) GCN layer 1
  matmul48_kernel<<<NNODES / 64, 128, 0, stream>>>(state, g1w, xw);
  zero_kernel<<<(NNODES * KCH + 255) / 256, 256, 0, stream>>>(agg, NNODES * KCH);
  scatter_kernel<<<NEDGES / 256, 256, 0, stream>>>(ei, ew, dinv, xw, agg);
  finish1_kernel<<<(NNODES * KCH + 255) / 256, 256, 0, stream>>>(agg, xw, dinv, g1b);

  // 4) GCN layer 2 (only node 0's output feeds the decoder)
  matmul48_kernel<<<NNODES / 64, 128, 0, stream>>>(agg /*x1*/, g2w, xw /*xw2*/);
  zero_kernel<<<1, 64, 0, stream>>>(interact, 48);
  gather0_kernel<<<NEDGES / 256, 256, 0, stream>>>(ei, ew, dinv, xw, interact);

  // 5) B=1 encoders + decoder
  small_enc_kernel<<<1, 160, 0, stream>>>(past, future, cpw, cpb, cfw, cfb,
                                          pWih, pWhh, pbi, pbh,
                                          fWih, fWhh, fbi, fbh,
                                          target, sfut, pres0);
  decoder_kernel<<<1, 448, 0, stream>>>(interact, xw, dinv, g2b, target, sfut, pres0,
                                        dWih, dWhh, dbi, dbh, fcw, fcb, outf);
}
